// ASPP_44229573214331
// MI455X (gfx1250) — matrix operand order repaired and checked by reading, not compiled, not run on hardware
//
#include <hip/hip_runtime.h>
#include <hip/hip_bf16.h>

// ---------------------------------------------------------------------------
// Types for CDNA5 WMMA (wave32): D(f32 16x16) = A(bf16 16x32) x B(bf16 32x16) + C
// ---------------------------------------------------------------------------
typedef __attribute__((ext_vector_type(16))) __bf16        v16bf;
typedef __attribute__((ext_vector_type(8)))  float         v8f;
typedef __attribute__((ext_vector_type(4)))  unsigned int  u32x4;
typedef __attribute__((ext_vector_type(4)))  float         f32x4;
typedef __attribute__((ext_vector_type(2)))  unsigned int  u32x2;
// 2-float vector with 4-byte alignment: pair gathers may be only dword-aligned
typedef float f32x2a __attribute__((ext_vector_type(2), aligned(4)));

// Problem constants (from reference setup_inputs)
#define BATCH   2
#define CIN     256
#define COUT    256
#define HH      64
#define WW      64
#define HW      (HH * WW)          // 4096
#define NBR     4                  // branches
#define KTAPS   9                  // 3x3
#define KDIM    (KTAPS * CIN)      // 2304
// rates: 6,12,18,24 == 6*(idx+1)

// Workspace layout (bytes)
#define WBF_ELEMS  (NBR * COUT * KTAPS * CIN)            // 2,359,296 bf16
#define WBF_BYTES  (WBF_ELEMS * 2)                        // 4,718,592
#define OFF_ELEMS  (NBR * BATCH * 18 * HW)                // 589,824 f32
// total ws use: WBF_BYTES + OFF_ELEMS*4 = 7,077,888 B (~6.8 MB)

// f32 -> bf16 (RNE), two values packed into one dword (lo=a, hi=b)
__device__ __forceinline__ unsigned pack2bf(float a, float b) {
    unsigned ua = __float_as_uint(a);
    unsigned ub = __float_as_uint(b);
    unsigned ra = ua + 0x7FFFu + ((ua >> 16) & 1u);
    unsigned rb = ub + 0x7FFFu + ((ub >> 16) & 1u);
    return (ra >> 16) | (rb & 0xFFFF0000u);
}
__device__ __forceinline__ unsigned short f2bf(float f) {
    unsigned u = __float_as_uint(f);
    unsigned r = u + 0x7FFFu + ((u >> 16) & 1u);
    return (unsigned short)(r >> 16);
}

union ABPack { u32x4 q[2]; v16bf v; };

// ---------------------------------------------------------------------------
// Kernel 1: repack dweights [4][256][256][3][3] f32 -> bf16 [i][oc][k9][c]
// ---------------------------------------------------------------------------
__global__ void aspp_prep_weights(const float* __restrict__ dw,
                                  unsigned short* __restrict__ wbf) {
    int idx = blockIdx.x * 256 + threadIdx.x;
    if (idx >= WBF_ELEMS) return;
    int c    = idx & 255;
    int k9   = (idx >> 8) % KTAPS;
    int ocag = idx / (256 * KTAPS);        // i*256 + oc
    wbf[idx] = f2bf(dw[(size_t)ocag * KDIM + c * KTAPS + k9]);
}

// ---------------------------------------------------------------------------
// Kernel 2: offset conv (weights/bias/rate of j=(i-1)%4), 256->18 dilated 3x3,
// bias+ReLU. LDS: x halo tile per channel + 18x9 weight slice (broadcast).
// ---------------------------------------------------------------------------
__global__ void aspp_offset_conv(const float* __restrict__ x,
                                 const float* __restrict__ ow,
                                 const float* __restrict__ ob,
                                 float* __restrict__ off) {
    const int tid = threadIdx.x;
    const int i   = blockIdx.y >> 1;
    const int b   = blockIdx.y & 1;
    const int j   = (i + 3) & 3;           // (i-1) mod 4
    const int rate = 6 * (j + 1);
    const int ty  = (blockIdx.x >> 2) * 16;
    const int tx  = (blockIdx.x & 3) * 16;

    __shared__ float xs[64 * 65];
    __shared__ float wl[18 * 9];

    const int py = tid >> 4;
    const int px = tid & 15;

    float acc[18];
#pragma unroll
    for (int ch = 0; ch < 18; ++ch) acc[ch] = ob[j * 18 + ch];

    const int n = 16 + 2 * rate;           // halo tile edge (28/40/52/64)
    const float* xb = x + (size_t)b * CIN * HW;

    for (int c = 0; c < CIN; ++c) {
        __syncthreads();
        for (int k = tid; k < n * n; k += 256) {
            int ly = k / n, lx = k - ly * n;
            int gy = ty + ly - rate, gx = tx + lx - rate;
            float v = 0.f;
            if ((unsigned)gy < (unsigned)HH && (unsigned)gx < (unsigned)WW)
                v = xb[(size_t)c * HW + gy * WW + gx];
            xs[ly * 65 + lx] = v;
        }
        if (tid < 18 * 9) {
            int ch = tid / 9, t = tid - ch * 9;
            wl[tid] = ow[((size_t)(j * 18 + ch) * CIN + c) * 9 + t];
        }
        __syncthreads();
#pragma unroll
        for (int t = 0; t < 9; ++t) {
            float v = xs[(py + (t / 3) * rate) * 65 + (px + (t % 3) * rate)];
#pragma unroll
            for (int ch = 0; ch < 18; ++ch)
                acc[ch] = fmaf(wl[ch * 9 + t], v, acc[ch]);
        }
    }

    float* op = off + ((size_t)(i * BATCH + b) * 18) * HW + (ty + py) * WW + (tx + px);
#pragma unroll
    for (int ch = 0; ch < 18; ++ch)
        op[(size_t)ch * HW] = fmaxf(acc[ch], 0.f);
}

// ---------------------------------------------------------------------------
// Kernel 3: deformable conv as implicit GEMM on WMMA.
// Bilinear corners gathered as horizontally-adjacent PAIRS (global_load_b64):
// pair base bx = clamp(x0,0,62); the two x-weights are redistributed onto the
// loaded elements, exactly reproducing per-corner valid-mask semantics.
// Each thread builds 2 S rows x 2 adjacent channels (packed b32 LDS stores).
// ---------------------------------------------------------------------------
__global__ void __launch_bounds__(256)
aspp_deform_wmma(const float* __restrict__ x,
                 const unsigned short* __restrict__ wbf,
                 const float* __restrict__ off,
                 float* __restrict__ out) {
    const int tid  = threadIdx.x;
    const int lane = tid & 31;
    const int wid  = tid >> 5;
    const int wm   = wid & 3;              // M sub-tile (64 oc)
    const int wn   = wid >> 2;             // N sub-tile (16 pixels)
    const int mrow = lane & 15;
    const int khalf = lane >> 4;

    const int i = blockIdx.y >> 1;
    const int b = blockIdx.y & 1;
    const int rate = 6 * (i + 1);          // deform taps use RATES[i]
    const int row = blockIdx.x >> 1;
    const int colBase = (blockIdx.x & 1) * 32;

    // Double-buffered S tile: 32 pixels x 32 channels bf16; pitch 40 ushorts
    __shared__ __align__(16) unsigned short Slds[2][32 * 40];
    __shared__ __align__(8)  int   sIdxB[32][2];   // byte offsets: top/bot pair
    __shared__ __align__(16) float sW[32][4];      // wt0, wt1, wb0, wb1

    const float* offp = off + (size_t)(i * BATCH + b) * 18 * HW;
    const char*  xb   = (const char*)(x + (size_t)b * CIN * HW);  // uniform base
    const char*  wb   = (const char*)wbf;                         // uniform base

    // S-build mapping: thread owns rows prow & prow+16, channels c2 & c2+1
    const int prow = tid >> 4;                         // 0..15
    const int c2   = (tid & 15) * 2;                   // even channel
    const unsigned c2B = (unsigned)c2 * (HW * 4);      // channel byte offset

    v8f acc[4];
#pragma unroll
    for (int mt = 0; mt < 4; ++mt)
#pragma unroll
        for (int r = 0; r < 8; ++r) acc[mt][r] = 0.f;

    for (int k9 = 0; k9 < KTAPS; ++k9) {
        // ---- per-tap sampling params (one pixel per lane) ----
        if (tid < 32) {
            int p = tid, col = colBase + p;
            float dy = offp[(size_t)(2 * k9) * HW + row * WW + col];
            float dx = offp[(size_t)(2 * k9 + 1) * HW + row * WW + col];
            float py = (float)row + (float)((k9 / 3 - 1) * rate) + dy;
            float px = (float)col + (float)((k9 % 3 - 1) * rate) + dx;
            float fy = floorf(py), fx = floorf(px);
            float wy = py - fy, wx = px - fx;
            int y0 = (int)fy, x0 = (int)fx;

            // x pair: base bx in [0,62]; element weights we0/we1
            int bx = min(max(x0, 0), WW - 2);
            float we0 = (bx == x0) ? (1.f - wx) : ((bx == x0 + 1) ? wx : 0.f);
            float we1 = (bx + 1 == x0) ? (1.f - wx) : ((bx + 1 == x0 + 1) ? wx : 0.f);
            // y rows: validity folded into weights
            bool vy0 = (y0 >= 0) & (y0 < HH);
            bool vy1 = (y0 >= -1) & (y0 < HH - 1);
            int yt = min(max(y0, 0), HH - 1);
            int yb = min(max(y0 + 1, 0), HH - 1);
            sIdxB[p][0] = (yt * WW + bx) * 4;
            sIdxB[p][1] = (yb * WW + bx) * 4;
            sW[p][0] = vy0 ? (1.f - wy) * we0 : 0.f;
            sW[p][1] = vy0 ? (1.f - wy) * we1 : 0.f;
            sW[p][2] = vy1 ? wy * we0 : 0.f;
            sW[p][3] = vy1 ? wy * we1 : 0.f;
        }
        __syncthreads();

        // ---- register-cache params for this thread's 2 S rows ----
        unsigned rIT[2], rIB[2];
        float    rW[2][4];
#pragma unroll
        for (int m = 0; m < 2; ++m) {
            int p = prow + 16 * m;
            u32x2 qi = *(const u32x2*)sIdxB[p];
            f32x4 qw = *(const f32x4*)sW[p];
            rIT[m] = qi[0];
            rIB[m] = qi[1];
#pragma unroll
            for (int k = 0; k < 4; ++k) rW[m][k] = qw[k];
        }

        // S-tile builder: 2 rows x 2 channels; 8 paired b64 gathers / chunk
        auto buildS = [&](int chunk, int buf) {
            unsigned cb0 = c2B + (unsigned)chunk * (32u * HW * 4u);
            unsigned cb1 = cb0 + (HW * 4u);
#pragma unroll
            for (int m = 0; m < 2; ++m) {
                f32x2a t0 = *(const f32x2a*)(xb + (cb0 + rIT[m]));
                f32x2a b0 = *(const f32x2a*)(xb + (cb0 + rIB[m]));
                f32x2a t1 = *(const f32x2a*)(xb + (cb1 + rIT[m]));
                f32x2a b1 = *(const f32x2a*)(xb + (cb1 + rIB[m]));
                float v0 = rW[m][0] * t0.x + rW[m][1] * t0.y
                         + rW[m][2] * b0.x + rW[m][3] * b0.y;
                float v1 = rW[m][0] * t1.x + rW[m][1] * t1.y
                         + rW[m][2] * b1.x + rW[m][3] * b1.y;
                *(unsigned*)&Slds[buf][(prow + 16 * m) * 40 + c2] = pack2bf(v0, v1);
            }
        };

        buildS(0, 0);
        __syncthreads();

#pragma unroll
        for (int idx = 0; idx < 8; ++idx) {
            if (idx < 7) buildS(idx + 1, (idx + 1) & 1);  // overlap next chunk

            // ---- B operand: 32x16 bf16 from LDS (contiguous k = 16*khalf + s, same order as A) ----
            ABPack bp;
            {
                const u32x4* q =
                    (const u32x4*)&Slds[idx & 1][(wn * 16 + mrow) * 40 + khalf * 16];
                bp.q[0] = q[0];
                bp.q[1] = q[1];
            }

            // ---- A operands + WMMA: 4 M-tiles per wave ----
#pragma unroll
            for (int mt = 0; mt < 4; ++mt) {
                int oc = wm * 64 + mt * 16 + mrow;
                unsigned abase =
                    ((unsigned)((i * COUT + oc) * KTAPS + k9) * CIN + (unsigned)(idx * 32))
                        * 2u + (unsigned)(khalf * 32);
                ABPack apk;
                apk.q[0] = *(const u32x4*)(wb + abase);
                apk.q[1] = *(const u32x4*)(wb + abase + 16);
                acc[mt] = __builtin_amdgcn_wmma_f32_16x16x32_bf16(
                    false, apk.v, false, bp.v, (short)0, acc[mt], false, false);
            }
            __syncthreads();
        }
    }

    // ---- store: D layout -> lane elem r is (M = r + 8*khalf, N = lane&15) ----
    const int col = colBase + wn * 16 + mrow;
    char* obp = (char*)out;                // uniform base, u32 voffsets
#pragma unroll
    for (int mt = 0; mt < 4; ++mt) {
#pragma unroll
        for (int r = 0; r < 8; ++r) {
            int oc = wm * 64 + mt * 16 + r + 8 * khalf;
            unsigned o = ((unsigned)(b * (NBR * COUT) + i * COUT + oc) * HW
                          + (unsigned)(row * WW + col)) * 4u;
            *(float*)(obp + o) = acc[mt][r];
        }
    }
}

// ---------------------------------------------------------------------------
// Launch
// ---------------------------------------------------------------------------
extern "C" void kernel_launch(void* const* d_in, const int* in_sizes, int n_in,
                              void* d_out, int out_size, void* d_ws, size_t ws_size,
                              hipStream_t stream) {
    const float* x  = (const float*)d_in[0];   // [2,256,64,64]
    const float* dw = (const float*)d_in[1];   // [4,256,256,3,3]
    const float* ow = (const float*)d_in[2];   // [4,18,256,3,3]
    const float* ob = (const float*)d_in[3];   // [4,18]
    float* out = (float*)d_out;                // [2,1024,64,64]

    unsigned short* wbf = (unsigned short*)d_ws;                    // bf16 weights
    float* offws = (float*)((char*)d_ws + WBF_BYTES);               // offset fields

    // 1) repack deform weights to bf16 [i][oc][k9][c]
    aspp_prep_weights<<<(WBF_ELEMS + 255) / 256, 256, 0, stream>>>(dw, wbf);

    // 2) offset convs (all 4 branches x 2 batches), LDS-tiled direct conv
    aspp_offset_conv<<<dim3(16, NBR * BATCH), 256, 0, stream>>>(x, ow, ob, offws);

    // 3) deformable conv implicit GEMM on WMMA bf16: grid = (128 strips, 8)
    aspp_deform_wmma<<<dim3(HW / 32, NBR * BATCH), 256, 0, stream>>>(
        x, wbf, offws, out);
}
